// BiLSTMDualPathwayCRF_82995948028540
// MI455X (gfx1250) — matrix operand order repaired and checked by reading, not compiled, not run on hardware
//
#include <hip/hip_runtime.h>

typedef __attribute__((ext_vector_type(16))) _Float16 v16h;
typedef __attribute__((ext_vector_type(8)))  _Float16 v8h;
typedef __attribute__((ext_vector_type(8)))  float    v8f;

#define NB   64
#define NT   512
#define NK   15
#define NTOK (NB * NT)   // 32768
#define GATES 1024       // 4H
#define HID_ 256

#if defined(__HIP_DEVICE_COMPILE__) && __has_builtin(__builtin_amdgcn_global_load_async_to_lds_b128)
#define HAVE_ASYNC 1
#else
#define HAVE_ASYNC 0
#endif

#if HAVE_ASYNC
typedef __attribute__((ext_vector_type(4))) int v4i;
// AS1 (global) pointers are 64-bit; AS3 (LDS) pointers are 32-bit and the low
// 32 bits of a generic LDS address are the LDS offset (flat->LDS truncation).
#define AS1P(p) ((__attribute__((address_space(1))) v4i*)(unsigned long long)(p))
#define AS3P(p) ((__attribute__((address_space(3))) v4i*)(unsigned int)(unsigned long long)(p))
#if __has_builtin(__builtin_amdgcn_s_wait_asynccnt)
#define WAIT_ASYNC(n) __builtin_amdgcn_s_wait_asynccnt(n)
#else
#define WAIT_ASYNC(n) asm volatile("s_wait_asynccnt %0" :: "i"(n) : "memory")
#endif
#endif

static __device__ __forceinline__ v8f wmma_f16(v16h a, v16h b, v8f c) {
    return __builtin_amdgcn_wmma_f32_16x16x32_f16(false, a, false, b, (short)0, c, false, false);
}
static __device__ __forceinline__ float sigm(float x) { return 1.0f / (1.0f + __expf(-x)); }

// ---------------------------------------------------------------------------
// Char-embedding gather straight to f16: X[n,d] = emb[ids[n], d]
// ---------------------------------------------------------------------------
__global__ void gather_embed(const int* __restrict__ ids, const float* __restrict__ emb,
                             _Float16* __restrict__ X) {
    long i = (long)blockIdx.x * blockDim.x + threadIdx.x;
    if (i >= (long)NTOK * 128) return;
    int n = (int)(i >> 7), d = (int)(i & 127);
    X[i] = (_Float16)emb[(long)ids[n] * 128 + d];
}

// f32 -> f16 conversion (weights + word embeddings, done once per launch)
__global__ void cvt_f16(const float* __restrict__ in, _Float16* __restrict__ out, long n) {
    long i = (long)blockIdx.x * blockDim.x + threadIdx.x;
    if (i < n) out[i] = (_Float16)in[i];
}

// cls_w2 (15,512) -> zero-padded (64,512) f16
__global__ void cvt_pad_w2(const float* __restrict__ in, _Float16* __restrict__ out) {
    int i = blockIdx.x * blockDim.x + threadIdx.x;   // 64*512
    if (i >= 64 * 512) return;
    int r = i >> 9, c = i & 511;
    out[i] = (r < NK) ? (_Float16)in[r * 512 + c] : (_Float16)0.0f;
}

// ---------------------------------------------------------------------------
// WMMA GEMM, all-f16 operands:  C[m,n] = sum_k A[m,k]*W[n,k] + b0[n] + b1[n]
// A = virtual concat of A0 (k<asplit) | A1. Double-buffered LDS tiles filled
// via async global->LDS b128 copies (ASYNCcnt) when available.
// Block 64x64, 8 waves x (1 row-block, 2 col-blocks). z batches directions.
// tmajor: permute store rows from token-major (b*T+t) to time-major (t*64+b).
// ---------------------------------------------------------------------------
__global__ void __launch_bounds__(256)
wmma_gemm(const _Float16* __restrict__ A0, int lda0,
          const _Float16* __restrict__ A1, int lda1, int asplit,
          const _Float16* __restrict__ W, long wz,
          const float* __restrict__ b0, const float* __restrict__ b1, int bz,
          void* __restrict__ Cp, int ldc, long cz,
          int Kdim, int Nact, int relu, int c_f16, int tmajor)
{
    const int zb = blockIdx.z;
    W += (long)zb * wz;
    if (b0) b0 += (long)zb * bz;
    if (b1) b1 += (long)zb * bz;

    const int m0 = blockIdx.y * 64, n0 = blockIdx.x * 64;
    __shared__ _Float16 As[2][64][40];   // pitch 40 halves = 80B (16B multiple)
    __shared__ _Float16 Bs[2][64][40];

    const int tid  = threadIdx.x;
    const int lane = tid & 31, wave = tid >> 5;
    const int hlf  = lane >> 4, l15 = lane & 15;
    const int wr   = wave >> 1;           // row block 0..3
    const int wc0  = (wave & 1) * 2;      // col blocks wc0, wc0+1

    // staging assignment: each thread copies one 16B chunk of A and one of B
    const int sr = tid >> 2, sc = tid & 3;        // row 0..63, chunk 0..3

    auto issue = [&](int buf, int k0) {
        const int kk = k0 + sc * 8;
        const _Float16* ap = (kk < asplit)
            ? A0 + (long)(m0 + sr) * lda0 + kk
            : A1 + (long)(m0 + sr) * lda1 + (kk - asplit);
        const _Float16* bp = W + (long)(n0 + sr) * Kdim + kk;
#if HAVE_ASYNC
        __builtin_amdgcn_global_load_async_to_lds_b128(
            AS1P(ap), AS3P(&As[buf][sr][sc * 8]), 0, 0);
        __builtin_amdgcn_global_load_async_to_lds_b128(
            AS1P(bp), AS3P(&Bs[buf][sr][sc * 8]), 0, 0);
#else
        *(uint4*)&As[buf][sr][sc * 8] = *(const uint4*)ap;
        *(uint4*)&Bs[buf][sr][sc * 8] = *(const uint4*)bp;
#endif
    };

    v8f acc0 = {}, acc1 = {};
    const int nkt = Kdim >> 5;
    issue(0, 0);

    for (int kb = 0; kb < nkt; ++kb) {
        const int cur = kb & 1;
        if (kb + 1 < nkt) issue(cur ^ 1, (kb + 1) << 5);
#if HAVE_ASYNC
        if (kb + 1 < nkt) { WAIT_ASYNC(2); } else { WAIT_ASYNC(0); }
#endif
        __syncthreads();   // current tile complete in LDS for all waves

        v16h af;
        {
            v8h lo = *(const v8h*)&As[cur][16 * wr + l15][16 * hlf];
            v8h hi = *(const v8h*)&As[cur][16 * wr + l15][16 * hlf + 8];
            #pragma unroll
            for (int i = 0; i < 8; ++i) { af[i] = lo[i]; af[i + 8] = hi[i]; }
        }
        v16h bf0, bf1;
        {
            const v8h* q = (const v8h*)&Bs[cur][16 * wc0 + l15][16 * hlf];
            v8h lo = q[0], hi = q[1];
            #pragma unroll
            for (int i = 0; i < 8; ++i) { bf0[i] = lo[i]; bf0[i + 8] = hi[i]; }
        }
        {
            const v8h* q = (const v8h*)&Bs[cur][16 * (wc0 + 1) + l15][16 * hlf];
            v8h lo = q[0], hi = q[1];
            #pragma unroll
            for (int i = 0; i < 8; ++i) { bf1[i] = lo[i]; bf1[i + 8] = hi[i]; }
        }
        acc0 = wmma_f16(af, bf0, acc0);
        acc1 = wmma_f16(af, bf1, acc1);
        __syncthreads();   // reads done before buffer reuse
    }

    // store: element v -> m = 16*wr + v + 8*hlf, n = 16*cb + l15
    #pragma unroll
    for (int tt = 0; tt < 2; ++tt) {
        const v8f acc = tt ? acc1 : acc0;
        const int gn = n0 + 16 * (wc0 + tt) + l15;
        if (gn < Nact) {
            float bias = (b0 ? b0[gn] : 0.0f) + (b1 ? b1[gn] : 0.0f);
            #pragma unroll
            for (int v = 0; v < 8; ++v) {
                int gm = m0 + 16 * wr + v + 8 * hlf;
                float val = acc[v] + bias;
                if (relu) val = fmaxf(val, 0.0f);
                long row = tmajor ? (long)((gm & (NT - 1)) * NB + (gm >> 9))
                                  : (long)gm;
                long idx = (long)zb * cz + row * ldc + gn;
                if (c_f16) ((_Float16*)Cp)[idx] = (_Float16)val;
                else       ((float*)Cp)[idx]     = val;
            }
        }
    }
}

// ---------------------------------------------------------------------------
// Persistent LSTM time scan: one workgroup per direction (1024 thr = 32 waves).
// gates(64x1024) = xg[t] + h_prev(64x256) @ Whh^T via WMMA; h in LDS f16,
// c in registers. xg is time-major: [t][b][gate] -> 256KB/step streamed.
// ---------------------------------------------------------------------------
__global__ void __launch_bounds__(1024)
lstm_scan(const float* __restrict__ xg,      // (2, T, B, 1024) time-major
          const _Float16* __restrict__ whh,  // (2, 1024, 256) f16
          _Float16* __restrict__ out)        // (NTOK, 512); dir -> cols dir*256..
{
    const int dir = blockIdx.x;
    xg  += (long)dir * ((long)NTOK * GATES);
    whh += (long)dir * ((long)GATES * HID_);

    __shared__ _Float16 hbuf[64][264];   // pitch 528B (33*16B)

    const int tid  = threadIdx.x;
    const int lane = tid & 31, wave = tid >> 5;
    const int hlf  = lane >> 4, l15 = lane & 15;
    const int wr   = wave >> 3;           // 0..3
    const int wc0  = (2 * wave) & 15;     // col blocks wc0, wc0+1 (of 16)

    for (int i = tid; i < 64 * 264; i += 1024) ((_Float16*)hbuf)[i] = (_Float16)0.0f;
    float cst[2][8] = {};
    __syncthreads();

    for (int s = 0; s < NT; ++s) {
        const int t = dir ? (NT - 1 - s) : s;
        const float* xgt = xg + (long)t * NB * GATES;

        v8f g[2][4];
        #pragma unroll
        for (int tt = 0; tt < 2; ++tt) {
            int gnb = (wc0 + tt) * 16 + l15;
            #pragma unroll
            for (int gg = 0; gg < 4; ++gg) {
                int gn = gg * HID_ + gnb;
                #pragma unroll
                for (int v = 0; v < 8; ++v) {
                    int b = 16 * wr + v + 8 * hlf;
                    g[tt][gg][v] = xgt[(long)b * GATES + gn];
                }
            }
        }

        for (int k0 = 0; k0 < HID_; k0 += 32) {
            v16h af;
            {
                v8h lo = *(const v8h*)&hbuf[16 * wr + l15][k0 + 16 * hlf];
                v8h hi = *(const v8h*)&hbuf[16 * wr + l15][k0 + 16 * hlf + 8];
                #pragma unroll
                for (int i = 0; i < 8; ++i) { af[i] = lo[i]; af[i + 8] = hi[i]; }
            }
            #pragma unroll
            for (int tt = 0; tt < 2; ++tt) {
                #pragma unroll
                for (int gg = 0; gg < 4; ++gg) {
                    int grow = gg * HID_ + (wc0 + tt) * 16 + l15;
                    const v8h* q = (const v8h*)&whh[(long)grow * HID_ + k0 + 16 * hlf];
                    v8h lo = q[0], hi = q[1];
                    v16h bf;
                    #pragma unroll
                    for (int i = 0; i < 8; ++i) { bf[i] = lo[i]; bf[i + 8] = hi[i]; }
                    g[tt][gg] = wmma_f16(af, bf, g[tt][gg]);
                }
            }
        }
        __syncthreads();   // all h_prev reads complete

        #pragma unroll
        for (int tt = 0; tt < 2; ++tt) {
            int j = (wc0 + tt) * 16 + l15;
            #pragma unroll
            for (int v = 0; v < 8; ++v) {
                int b = 16 * wr + v + 8 * hlf;
                float iv = sigm(g[tt][0][v]);
                float fv = sigm(g[tt][1][v]);
                float gv = tanhf(g[tt][2][v]);
                float ov = sigm(g[tt][3][v]);
                float c  = fv * cst[tt][v] + iv * gv;
                cst[tt][v] = c;
                float h = ov * tanhf(c);
                hbuf[b][j] = (_Float16)h;
                out[((long)b * NT + t) * 512 + dir * HID_ + j] = (_Float16)h;
            }
        }
        __syncthreads();
    }
}

// ---------------------------------------------------------------------------
// CRF NLL (K=15), 64 lanes = 64 batch rows; logits pitch 16, col 15 ignored.
// ---------------------------------------------------------------------------
__global__ void __launch_bounds__(64)
crf_nll(const float* __restrict__ logits, const int* __restrict__ tags,
        const float* __restrict__ start, const float* __restrict__ endw,
        const float* __restrict__ trans, float* __restrict__ outv)
{
    __shared__ float tr[NK][NK];
    __shared__ float st[NK], en[NK];
    __shared__ float red[64];
    const int tid = threadIdx.x;
    if (tid < NK) { st[tid] = start[tid]; en[tid] = endw[tid]; }
    for (int i = tid; i < NK * NK; i += 64) tr[i / NK][i % NK] = trans[i];
    __syncthreads();

    const float* em = logits + (long)tid * NT * 16;
    const int*   tg = tags   + (long)tid * NT;

    float alpha[NK];
    int tp = tg[0];
    float num = st[tp] + em[tp];
    #pragma unroll
    for (int j = 0; j < NK; ++j) alpha[j] = st[j] + em[j];

    for (int t = 1; t < NT; ++t) {
        int tc = tg[t];
        num += tr[tp][tc] + em[t * 16 + tc];
        tp = tc;
        float na[NK];
        #pragma unroll
        for (int j = 0; j < NK; ++j) {
            float m = -3.4e38f;
            #pragma unroll
            for (int i = 0; i < NK; ++i) m = fmaxf(m, alpha[i] + tr[i][j]);
            float s = 0.0f;
            #pragma unroll
            for (int i = 0; i < NK; ++i) s += __expf(alpha[i] + tr[i][j] - m);
            na[j] = m + __logf(s) + em[t * 16 + j];
        }
        #pragma unroll
        for (int j = 0; j < NK; ++j) alpha[j] = na[j];
    }
    num += en[tp];
    float m = -3.4e38f;
    #pragma unroll
    for (int j = 0; j < NK; ++j) m = fmaxf(m, alpha[j] + en[j]);
    float s = 0.0f;
    #pragma unroll
    for (int j = 0; j < NK; ++j) s += __expf(alpha[j] + en[j] - m);
    red[tid] = num - (m + __logf(s));
    __syncthreads();
    if (tid == 0) {
        float a = 0.0f;
        for (int i = 0; i < 64; ++i) a += red[i];
        outv[0] = -a / 64.0f;
    }
}

// ---------------------------------------------------------------------------
extern "C" void kernel_launch(void* const* d_in, const int* in_sizes, int n_in,
                              void* d_out, int out_size, void* d_ws, size_t ws_size,
                              hipStream_t stream) {
    const int*   char_ids = (const int*)  d_in[0];
    const float* word_emb = (const float*)d_in[1];
    const int*   tags     = (const int*)  d_in[2];
    const float* emb_tab  = (const float*)d_in[3];
    const float* Wih[4] = { (const float*)d_in[4],  (const float*)d_in[8],
                            (const float*)d_in[12], (const float*)d_in[16] };
    const float* Whh[4] = { (const float*)d_in[5],  (const float*)d_in[9],
                            (const float*)d_in[13], (const float*)d_in[17] };
    const float* Bih[4] = { (const float*)d_in[6],  (const float*)d_in[10],
                            (const float*)d_in[14], (const float*)d_in[18] };
    const float* Bhh[4] = { (const float*)d_in[7],  (const float*)d_in[11],
                            (const float*)d_in[15], (const float*)d_in[19] };
    const float* cls_w1 = (const float*)d_in[20];
    const float* cls_b1 = (const float*)d_in[21];
    const float* cls_w2 = (const float*)d_in[22];
    const float* cls_b2 = (const float*)d_in[23];
    const float* crf_s  = (const float*)d_in[24];
    const float* crf_e  = (const float*)d_in[25];
    const float* crf_tr = (const float*)d_in[26];

    // ---- workspace layout (fp32 first, then f16) ----
    char* ws = (char*)d_ws;
    float* XG   = (float*)ws;                  ws += 2L * NTOK * GATES * 4;   // 256MB
    float* LOGI = (float*)ws;                  ws += (long)NTOK * 16 * 4;     // 2MB
    _Float16* Xc16   = (_Float16*)ws;          ws += (long)NTOK * 128 * 2;
    _Float16* We16   = (_Float16*)ws;          ws += (long)NTOK * 768 * 2;
    _Float16* CH0    = (_Float16*)ws;          ws += (long)NTOK * 512 * 2;
    _Float16* CH1    = (_Float16*)ws;          ws += (long)NTOK * 512 * 2;
    _Float16* WH0    = (_Float16*)ws;          ws += (long)NTOK * 512 * 2;
    _Float16* WH1    = (_Float16*)ws;          ws += (long)NTOK * 512 * 2;
    _Float16* HID16  = (_Float16*)ws;          ws += (long)NTOK * 512 * 2;
    _Float16* WHH16  = (_Float16*)ws;          ws += 4L * 2 * GATES * HID_ * 2;
    _Float16* W1_16  = (_Float16*)ws;          ws += 512L * 1024 * 2;
    _Float16* W2_16  = (_Float16*)ws;          ws += 64L * 512 * 2;
    _Float16* WIH16  = (_Float16*)ws;          // 2*1024*(128+512+768+512) halves

    const int  kin[4]   = { 128, 512, 768, 512 };
    const long whh_n    = 2L * GATES * HID_;
    long wih_off[4]; { long o = 0; for (int l = 0; l < 4; ++l) { wih_off[l] = o; o += 2L * GATES * kin[l]; } }

    // ---- one-time conversions ----
    gather_embed<<<(NTOK * 128 + 255) / 256, 256, 0, stream>>>(char_ids, emb_tab, Xc16);
    cvt_f16<<<(int)(((long)NTOK * 768 + 255) / 256), 256, 0, stream>>>(word_emb, We16, (long)NTOK * 768);
    for (int l = 0; l < 4; ++l) {
        cvt_f16<<<(int)((whh_n + 255) / 256), 256, 0, stream>>>(Whh[l], WHH16 + l * whh_n, whh_n);
        long n = 2L * GATES * kin[l];
        cvt_f16<<<(int)((n + 255) / 256), 256, 0, stream>>>(Wih[l], WIH16 + wih_off[l], n);
    }
    cvt_f16<<<(512 * 1024 + 255) / 256, 256, 0, stream>>>(cls_w1, W1_16, 512L * 1024);
    cvt_pad_w2<<<(64 * 512 + 255) / 256, 256, 0, stream>>>(cls_w2, W2_16);

    const long xg_z = (long)NTOK * GATES;
    const _Float16* lin[4]  = { Xc16, CH0, We16, WH0 };
    _Float16*       lout[4] = { CH0, CH1, WH0, WH1 };

    for (int l = 0; l < 4; ++l) {
        // XG[dir] = X @ Wih[dir]^T + bih + bhh   (time-major store)
        wmma_gemm<<<dim3(GATES / 64, NTOK / 64, 2), 256, 0, stream>>>(
            lin[l], kin[l], lin[l], kin[l], kin[l],
            WIH16 + wih_off[l], (long)GATES * kin[l], Bih[l], Bhh[l], GATES,
            XG, GATES, xg_z, kin[l], GATES, 0, /*c_f16=*/0, /*tmajor=*/1);
        lstm_scan<<<dim3(2), dim3(1024), 0, stream>>>(XG, WHH16 + l * whh_n, lout[l]);
    }

    // classifier layer 1: relu([CH1 | WH1] @ cls_w1^T + b1) -> f16
    wmma_gemm<<<dim3(512 / 64, NTOK / 64, 1), 256, 0, stream>>>(
        CH1, 512, WH1, 512, 512,
        W1_16, 0, cls_b1, nullptr, 0,
        HID16, 512, 0, 1024, 512, 1, /*c_f16=*/1, /*tmajor=*/0);

    // classifier layer 2: logits (N=15, weights padded to 64 rows, pitch 16)
    wmma_gemm<<<dim3(1, NTOK / 64, 1), 256, 0, stream>>>(
        HID16, 512, HID16, 512, 512,
        W2_16, 0, cls_b2, nullptr, 0,
        LOGI, 16, 0, 512, NK, 0, /*c_f16=*/0, /*tmajor=*/0);

    crf_nll<<<1, 64, 0, stream>>>(LOGI, tags, crf_s, crf_e, crf_tr, (float*)d_out);
}
